// CausalSelfAttention_54365696033478
// MI455X (gfx1250) — matrix operand order repaired and checked by reading, not compiled, not run on hardware
//
#include <hip/hip_runtime.h>
#include <hip/hip_bf16.h>
#include <math.h>
#include <stdint.h>

typedef __attribute__((ext_vector_type(16))) _Float16 v16h;
typedef __attribute__((ext_vector_type(8)))  float    v8f;
typedef __attribute__((ext_vector_type(4)))  unsigned uint4v;

#define B_ 4
#define T_ 2048
#define E_ 1024
#define H_ 16
#define D_ 64

union AF   { v16h h; _Float16 e[16]; unsigned u[8]; };
union CF   { v8f v; float f[8]; };
union LD16 { uint4v q; _Float16 e[8]; };

// ---- CDNA5 async global->LDS copy (ASYNCcnt path), 16B per active lane ----
__device__ __forceinline__ void async_copy_b128(void* lds_ptr, const void* gptr) {
  const unsigned lds_off = (unsigned)(uintptr_t)lds_ptr;         // low 32b of generic
  const unsigned long long ga = (unsigned long long)(uintptr_t)gptr;
  asm volatile("global_load_async_to_lds_b128 %0, %1, off"
               :: "v"(lds_off), "v"(ga) : "memory");
}
__device__ __forceinline__ void wait_async0() {
  asm volatile("s_wait_asynccnt 0" ::: "memory");
}

// A-fragment (16x32 f16, MxK). ISA layout: lane M = lane&15;
// lanes 0-15 hold K pairs {0..7, 16..23}; lanes 16-31 hold {8..15, 24..31}.
__device__ inline v16h load_a_frag(const _Float16* base, int stride, int lane) {
  const int m  = lane & 15;
  const int kw = (lane < 16) ? 0 : 4;            // dword offset = kb/2
  const unsigned* rp = (const unsigned*)(base + (size_t)m * stride);
  AF f;
#pragma unroll
  for (int r = 0; r < 4; ++r) {
    f.u[r]     = rp[kw + r];
    f.u[4 + r] = rp[8 + kw + r];
  }
  return f.h;
}

// B-fragment (32x16 f16, KxN) read from an N-major tile (row = N, contiguous K):
// lane N = lane&15; lanes 0-15 hold K {0..7, 16..23}; lanes 16-31 hold {8..15, 24..31}.
__device__ inline v16h load_b_frag(const _Float16* base, int stride, int lane, int koff) {
  const int n  = lane & 15;
  const int kb = (lane < 16) ? 0 : 8;
  const unsigned* rp = (const unsigned*)(base + (size_t)n * stride + koff + kb);
  AF f;
#pragma unroll
  for (int r = 0; r < 4; ++r) { f.u[r] = rp[r]; f.u[4 + r] = rp[8 + r]; }
  return f.h;
}

__device__ inline v8f wmma_f16(v16h a, v16h b, v8f c) {
  return __builtin_amdgcn_wmma_f32_16x16x32_f16(false, a, false, b, (short)0, c, false, false);
}

// ---------------------------------------------------------------------------
// GEMM: C[M,N] = A[M,K](f16) * B[K,N](f16) + bias(f32).
// 128x128 block tile, 8 waves (2x4), wave tile 64x32, K-step 64, double-buffered
// LDS. Per iteration: issue async A-copy + B global loads for tile s+1, run the
// 16 WMMAs of tile s, THEN scatter B into LDS (so the loadcnt wait lands after
// compute, not before it).
// ---------------------------------------------------------------------------
template<bool OUT_F16>
__global__ __launch_bounds__(256) void gemm_wmma_kernel(
    const _Float16* __restrict__ A, const _Float16* __restrict__ Bm,
    const float* __restrict__ bias, void* __restrict__ Cout,
    int M, int N, int K)
{
  constexpr int LSA = 72, LSB = 72;               // halves; 144B rows (16B-aligned cols)
  __shared__ _Float16 Alds[2][128 * LSA];         // 128 x 64 (M x K), row-major
  __shared__ _Float16 Blds[2][128 * LSB];         // 64 x 128 (K x N) stored as [N][K]
  const int tid = threadIdx.x, lane = tid & 31, wave = tid >> 5;
  const int wm = wave >> 2, wn = wave & 3;
  const int bm = blockIdx.y * 128, bn = blockIdx.x * 128;

  CF acc[4][2];
  v8f zero = {};
#pragma unroll
  for (int i = 0; i < 4; ++i)
#pragma unroll
    for (int j = 0; j < 2; ++j) acc[i][j].v = zero;

  auto issueA = [&](int k0, int buf) {            // async DMA, 16B per lane
#pragma unroll
    for (int j = 0; j < 4; ++j) {
      const int idx = tid * 32 + j * 8;
      const int row = idx >> 6, col = idx & 63;
      async_copy_b128(&Alds[buf][row * LSA + col],
                      A + (size_t)(bm + row) * K + k0 + col);
    }
  };
  auto loadB = [&](int k0, LD16* tb) {            // global loads into registers
#pragma unroll
    for (int j = 0; j < 4; ++j) {
      const int idx = tid * 32 + j * 8;
      const int kr = idx >> 7, nc = idx & 127;
      tb[j].q = *(const uint4v*)(Bm + (size_t)(k0 + kr) * N + bn + nc);
    }
  };
  auto storeB = [&](const LD16* tb, int buf) {    // transpose-scatter [K][N] -> [N][K]
#pragma unroll
    for (int j = 0; j < 4; ++j) {
      const int idx = tid * 32 + j * 8;
      const int kr = idx >> 7, nc = idx & 127;
#pragma unroll
      for (int i = 0; i < 8; ++i) Blds[buf][(nc + i) * LSB + kr] = tb[j].e[i];
    }
  };

  const int nk = K / 64;
  { // prologue: stage tile 0
    LD16 tb[4];
    issueA(0, 0);
    loadB(0, tb);
    storeB(tb, 0);
  }
  for (int s = 0; s < nk; ++s) {
    wait_async0();                                // this wave's async copies landed
    __syncthreads();                              // all waves' stage(s) visible
    LD16 tb[4];
    const bool pre = (s + 1 < nk);
    if (pre) {                                    // issue next tile's fetches first
      issueA((s + 1) * 64, (s + 1) & 1);
      loadB((s + 1) * 64, tb);
    }

    const _Float16* Ab = Alds[s & 1] + (wm * 64) * LSA;
    const _Float16* Bb = Blds[s & 1] + (wn * 32) * LSB;
#pragma unroll
    for (int ks = 0; ks < 2; ++ks) {
      v16h af[4], bf[2];
#pragma unroll
      for (int i = 0; i < 4; ++i)
        af[i] = load_a_frag(Ab + i * 16 * LSA + ks * 32, LSA, lane);
#pragma unroll
      for (int j = 0; j < 2; ++j)
        bf[j] = load_b_frag(Bb + j * 16 * LSB, LSB, lane, ks * 32);
#pragma unroll
      for (int i = 0; i < 4; ++i)
#pragma unroll
        for (int j = 0; j < 2; ++j)
          acc[i][j].v = wmma_f16(af[i], bf[j], acc[i][j].v);
    }

    if (pre) storeB(tb, (s + 1) & 1);             // scatter after compute
  }

  // Epilogue: C layout -> row = r + 8*(lane>=16), col = lane&15
  const int nn = lane & 15, hi = lane >> 4;
#pragma unroll
  for (int i = 0; i < 4; ++i)
#pragma unroll
    for (int j = 0; j < 2; ++j) {
      const int n  = bn + wn * 32 + j * 16 + nn;
      const float bv = bias[n];
#pragma unroll
      for (int r = 0; r < 8; ++r) {
        const int m = bm + wm * 64 + i * 16 + r + 8 * hi;
        const float val = acc[i][j].f[r] + bv;
        if (OUT_F16) ((_Float16*)Cout)[(size_t)m * N + n] = (_Float16)val;
        else         ((float*)Cout)[(size_t)m * N + n] = val;
      }
    }
}

// ---------------------------------------------------------------------------
// Flash attention: one block = 128 q rows of one (b, h); 8 waves x 16 rows.
// K/V tiles double-buffered in LDS: tile kt+1's async K copy and V global loads
// are issued before tile kt's compute; V's transpose-scatter happens after.
// ---------------------------------------------------------------------------
__global__ __launch_bounds__(256) void flash_attn_kernel(
    const _Float16* __restrict__ qkv, _Float16* __restrict__ out)
{
  constexpr int LS = 72;
  __shared__ _Float16 Klds[2][64 * LS];           // [key][d]
  __shared__ _Float16 Vlds[2][64 * LS];           // [d][key] (transposed)
  __shared__ _Float16 Plds[8][16 * LS];           // per-wave P scratch [qrow][key]

  const int qblk = blockIdx.x, h = blockIdx.y, b = blockIdx.z;
  const int tid = threadIdx.x, lane = tid & 31, wave = tid >> 5;
  const int q0 = qblk * 128;
  const int nn = lane & 15, hi = lane >> 4;
  const int rstride = 3 * E_;

  auto issueK = [&](int kt, int buf) {            // verbatim tile copy -> async DMA
#pragma unroll
    for (int j = 0; j < 2; ++j) {
      const int idx = tid * 16 + j * 8;
      const int key = idx >> 6, d0 = idx & 63;
      async_copy_b128(&Klds[buf][key * LS + d0],
                      qkv + ((size_t)b * T_ + kt * 64 + key) * rstride + h * D_ + E_ + d0);
    }
  };
  auto loadV = [&](int kt, LD16* tv) {
#pragma unroll
    for (int j = 0; j < 2; ++j) {
      const int idx = tid * 16 + j * 8;
      const int key = idx >> 6, d0 = idx & 63;
      tv[j].q = *(const uint4v*)(qkv + ((size_t)b * T_ + kt * 64 + key) * rstride +
                                 h * D_ + 2 * E_ + d0);
    }
  };
  auto storeV = [&](const LD16* tv, int buf) {    // transpose-scatter [key][d] -> [d][key]
#pragma unroll
    for (int j = 0; j < 2; ++j) {
      const int idx = tid * 16 + j * 8;
      const int key = idx >> 6, d0 = idx & 63;
#pragma unroll
      for (int i = 0; i < 8; ++i) Vlds[buf][(d0 + i) * LS + key] = tv[j].e[i];
    }
  };

  // Persistent Q fragments (16 rows x 64 d = 2 A-frags) straight from global.
  const _Float16* qbase = qkv + ((size_t)b * T_ + q0 + wave * 16) * rstride + h * D_;
  v16h qf[2];
#pragma unroll
  for (int s = 0; s < 2; ++s) qf[s] = load_a_frag(qbase + s * 32, rstride, lane);

  CF o[4];
  float m_i[8], l_i[8];
  v8f zero = {};
#pragma unroll
  for (int d = 0; d < 4; ++d) o[d].v = zero;
#pragma unroll
  for (int r = 0; r < 8; ++r) { m_i[r] = -__builtin_inff(); l_i[r] = 0.f; }

  const int nkt = 2 * qblk + 2;                   // causal: keys 0 .. q0+127
  { // prologue: stage tile 0
    LD16 tv[2];
    issueK(0, 0);
    loadV(0, tv);
    storeV(tv, 0);
  }
  for (int kt = 0; kt < nkt; ++kt) {
    wait_async0();
    __syncthreads();                              // tile kt published; kt-1 reads done
    LD16 tv[2];
    const bool pre = (kt + 1 < nkt);
    if (pre) {                                    // issue next tile's fetches first
      issueK(kt + 1, (kt + 1) & 1);
      loadV(kt + 1, tv);
    }
    const _Float16* Kb = Klds[kt & 1];
    const _Float16* Vb = Vlds[kt & 1];

    // S = Q * K^T : 4 key-subtiles of 16, reduce over d=64 in 2 WMMA steps
    CF s[4];
#pragma unroll
    for (int ns = 0; ns < 4; ++ns) {
      v8f c = zero;
#pragma unroll
      for (int ks = 0; ks < 2; ++ks)
        c = wmma_f16(qf[ks], load_b_frag(Kb + ns * 16 * LS, LS, lane, ks * 32), c);
      s[ns].v = c;
    }

    // scale (1/sqrt(64) = 0.125) + causal mask
#pragma unroll
    for (int ns = 0; ns < 4; ++ns) {
      const int ki = kt * 64 + ns * 16 + nn;
#pragma unroll
      for (int r = 0; r < 8; ++r) {
        const int qi = q0 + wave * 16 + r + 8 * hi;
        const float sv = s[ns].f[r] * 0.125f;
        s[ns].f[r] = (ki <= qi) ? sv : -__builtin_inff();
      }
    }

    // online softmax (row reductions across the 16 lanes of each half-wave)
#pragma unroll
    for (int r = 0; r < 8; ++r) {
      float mloc = s[0].f[r];
#pragma unroll
      for (int ns = 1; ns < 4; ++ns) mloc = fmaxf(mloc, s[ns].f[r]);
#pragma unroll
      for (int off = 8; off >= 1; off >>= 1) mloc = fmaxf(mloc, __shfl_xor(mloc, off, 32));
      const float mnew  = fmaxf(m_i[r], mloc);
      const float alpha = __expf(m_i[r] - mnew);
      float psum = 0.f;
#pragma unroll
      for (int ns = 0; ns < 4; ++ns) {
        const float p = __expf(s[ns].f[r] - mnew);
        s[ns].f[r] = p; psum += p;
      }
#pragma unroll
      for (int off = 8; off >= 1; off >>= 1) psum += __shfl_xor(psum, off, 32);
      l_i[r] = l_i[r] * alpha + psum;
      m_i[r] = mnew;
#pragma unroll
      for (int d = 0; d < 4; ++d) o[d].f[r] *= alpha;
    }

    // P -> wave-private LDS (C-frag -> A-frag re-layout); intra-wave DS in-order
    _Float16* pw = &Plds[wave][0];
#pragma unroll
    for (int ns = 0; ns < 4; ++ns)
#pragma unroll
      for (int r = 0; r < 8; ++r)
        pw[(r + 8 * hi) * LS + ns * 16 + nn] = (_Float16)s[ns].f[r];

    // O += P * V : reduce over 64 keys (2 WMMA steps) for 4 d-subtiles
    v16h pf[2];
#pragma unroll
    for (int ks = 0; ks < 2; ++ks) pf[ks] = load_a_frag(pw + ks * 32, LS, lane);
#pragma unroll
    for (int dn = 0; dn < 4; ++dn)
#pragma unroll
      for (int ks = 0; ks < 2; ++ks)
        o[dn].v = wmma_f16(pf[ks], load_b_frag(Vb + dn * 16 * LS, LS, lane, ks * 32), o[dn].v);

    if (pre) storeV(tv, (kt + 1) & 1);            // scatter after compute
  }

  // epilogue: O / l -> f16 attention output [B,T,E]
#pragma unroll
  for (int r = 0; r < 8; ++r) {
    const float inv = 1.0f / l_i[r];
    const int row = q0 + wave * 16 + r + 8 * hi;
#pragma unroll
    for (int dn = 0; dn < 4; ++dn)
      out[((size_t)b * T_ + row) * E_ + h * D_ + dn * 16 + nn] =
          (_Float16)(o[dn].f[r] * inv);
  }
}

// ---------------------------------------------------------------------------
__global__ __launch_bounds__(256) void f32_to_f16_kernel(
    const float* __restrict__ in, _Float16* __restrict__ out, int n)
{
  const int i = (blockIdx.x * blockDim.x + threadIdx.x) * 4;
  if (i < n) {
    const float4 v = *(const float4*)(in + i);
    out[i + 0] = (_Float16)v.x; out[i + 1] = (_Float16)v.y;
    out[i + 2] = (_Float16)v.z; out[i + 3] = (_Float16)v.w;
  }
}

extern "C" void kernel_launch(void* const* d_in, const int* in_sizes, int n_in,
                              void* d_out, int out_size, void* d_ws, size_t ws_size,
                              hipStream_t stream) {
  const float* x      = (const float*)d_in[0];
  const float* w_qkv  = (const float*)d_in[1];
  const float* b_qkv  = (const float*)d_in[2];
  const float* w_proj = (const float*)d_in[3];
  const float* b_proj = (const float*)d_in[4];

  const size_t nX   = (size_t)B_ * T_ * E_;       //  8,388,608
  const size_t nWq  = (size_t)E_ * 3 * E_;        //  3,145,728
  const size_t nWp  = (size_t)E_ * E_;            //  1,048,576
  const size_t nQKV = (size_t)B_ * T_ * 3 * E_;   // 25,165,824

  // workspace layout (f16 halves), total ~92.3 MB; all slices 16B-aligned
  _Float16* x16   = (_Float16*)d_ws;
  _Float16* wq16  = x16   + nX;
  _Float16* wp16  = wq16  + nWq;
  _Float16* qkv16 = wp16  + nWp;
  _Float16* at16  = qkv16 + nQKV;

  f32_to_f16_kernel<<<(unsigned)(nX  / 1024), 256, 0, stream>>>(x,      x16,  (int)nX);
  f32_to_f16_kernel<<<(unsigned)(nWq / 1024), 256, 0, stream>>>(w_qkv,  wq16, (int)nWq);
  f32_to_f16_kernel<<<(unsigned)(nWp / 1024), 256, 0, stream>>>(w_proj, wp16, (int)nWp);

  // qkv = x @ w_qkv + b_qkv  (f16 out)
  gemm_wmma_kernel<true><<<dim3(3 * E_ / 128, B_ * T_ / 128), 256, 0, stream>>>(
      x16, wq16, b_qkv, (void*)qkv16, B_ * T_, 3 * E_, E_);

  // flash attention per (b, h, 128-row q block)
  flash_attn_kernel<<<dim3(T_ / 128, H_, B_), 256, 0, stream>>>(qkv16, at16);

  // out = attn @ w_proj + b_proj  (f32 out)
  gemm_wmma_kernel<false><<<dim3(E_ / 128, B_ * T_ / 128), 256, 0, stream>>>(
      at16, wp16, b_proj, d_out, B_ * T_, E_, E_);
}
